// EEGLSTMClassifier_21577915695124
// MI455X (gfx1250) — matrix operand order repaired and checked by reading, not compiled, not run on hardware
//
#include <hip/hip_runtime.h>

// ---------- types ----------
typedef __attribute__((ext_vector_type(16))) _Float16   v16h;
typedef __attribute__((ext_vector_type(8)))  float      v8f;
typedef __attribute__((ext_vector_type(4)))  unsigned   u32x4;

struct F16x16 { u32x4 lo, hi; };   // 32 bytes == v16h

__device__ inline unsigned short f2h(float f) {
  _Float16 h = (_Float16)f;
  return __builtin_bit_cast(unsigned short, h);
}
__device__ inline float sigmoidf(float x) {
  return 1.0f / (1.0f + __expf(-x));
}
// A fragment (16x32 f16): p must already include +half*8 elements
__device__ inline v16h ldfragA(const unsigned short* p) {
  F16x16 r;
  r.lo = *(const u32x4*)(p);
  r.hi = *(const u32x4*)(p + 16);
  return __builtin_bit_cast(v16h, r);
}
// B fragment (32x16 f16, packed fragment order): p = base + lane*16 + const
__device__ inline v16h ldfragB_lds(const unsigned short* p) {
  F16x16 r;
  r.lo = *(const u32x4*)(p);
  r.hi = *(const u32x4*)(p + 8);
  return __builtin_bit_cast(v16h, r);
}

// ---------- one-time weight pack into B-fragment layout ----------
// W_cat[n][k] = (k < in_dim) ? W_ih[n][k] : W_hh[n][k-in_dim], n in [0,2048), Ktot = 32*KT
// packed[((ntile*KT + kt)*32 + lane)*16 + j] = f16(W_cat[ntile*16 + lane%16][kt*32 + (lane/16)*8 + (j%8) + (j/8)*16])
__global__ void pack_weights(const float* __restrict__ Wih, const float* __restrict__ Whh,
                             int in_dim, int KT, unsigned short* __restrict__ dst) {
  int tid = blockIdx.x * blockDim.x + threadIdx.x;
  int total = 128 * KT * 32;
  if (tid >= total) return;
  int lane  = tid & 31;
  int rest  = tid >> 5;
  int kt    = rest % KT;
  int ntile = rest / KT;
  int n     = ntile * 16 + (lane & 15);
  int kbase = kt * 32 + (lane >> 4) * 8;
  unsigned short* d = dst + (((size_t)ntile * KT + kt) * 512 + (size_t)lane * 16);
  for (int j = 0; j < 16; ++j) {
    int k = kbase + (j & 7) + (j >> 3) * 16;
    float v = (k < in_dim) ? Wih[(size_t)n * in_dim + k]
                           : Whh[(size_t)n * 512 + (k - in_dim)];
    d[j] = f2h(v);
  }
}

// ---------- x [B,C,T] fp32 -> xin [t*B + b][c] f16 ----------
__global__ void x_to_f16(const float* __restrict__ x, unsigned short* __restrict__ xin) {
  int idx = blockIdx.x * blockDim.x + threadIdx.x;   // (t*64 + b)*64 + c
  if (idx >= 64 * 64 * 1000) return;
  int c = idx & 63;
  int b = (idx >> 6) & 63;
  int t = idx >> 12;
  xin[idx] = f2h(x[((size_t)b * 64 + c) * 1000 + t]);
}

// ---------- zero h double-buffer + barrier counter ----------
__global__ void scan_init(unsigned* __restrict__ hbuf_u32, unsigned* __restrict__ cnt) {
  int i = blockIdx.x * blockDim.x + threadIdx.x;     // 128*256 == 32768 dwords == 128KB
  if (i < 32768) hbuf_u32[i] = 0u;
  if (i == 0)    *cnt = 0u;
}

// ---------- persistent LSTM layer scan: 32 WGs x 128 threads ----------
// WG wg owns h columns [wg*16, wg*16+16) and gate columns {g*512 + wg*16 .. +16}, g=0..3.
// The WG's weight slice (4 gate n-tiles x KT k-tiles, fragment-packed) is staged in LDS
// ONCE, then each timestep reads B fragments from LDS (ds_load_b128) — weights stationary.
// Wave w = batch strip m (16 rows); 4 gate accumulators -> lane-local cell update.
// K loop fully unrolled (KT compile-time): fragment addresses are immediate offsets.
template <int KIN>
__global__ __launch_bounds__(128)
void lstm_scan(const unsigned short* __restrict__ xin,
               const unsigned short* __restrict__ packedW,
               const float* __restrict__ bih, const float* __restrict__ bhh,
               unsigned short* __restrict__ hseq,   // [T*64][512] f16 (next layer's input)
               unsigned short* __restrict__ hbuf,   // [2][64][512] f16 state, double buffered
               float* __restrict__ hT,              // [64][512] fp32, written at t==T-1
               unsigned* __restrict__ cnt, int T) {
  constexpr int KT = KIN + 16;        // total K tiles (Ktot = 32*KT)
  constexpr int IN = KIN * 32;        // input feature dim

  __shared__ unsigned short smem[4 * KT * 512];   // 4 gates x KT tiles x 1KB (<=128KB)

  const int wg   = blockIdx.x;        // 0..31
  const int tid  = threadIdx.x;       // 0..127
  const int lane = tid & 31;
  const int m    = tid >> 5;          // wave index == batch row strip
  const int ln   = lane & 15;
  const int half = lane >> 4;
  const int arow = m * 16 + ln;       // A-fragment batch row for this lane
  const int col  = wg * 16 + ln;      // h column this lane owns (C/D layout)

  // ---- stage this WG's weight slice into LDS (once; amortized over T steps) ----
  for (int idx = tid; idx < 4 * KT * 32; idx += 128) {
    int l   = idx & 31;
    int tl  = idx >> 5;               // g*KT + kt
    int kt  = tl % KT;
    int g   = tl / KT;
    int ntile = g * 32 + wg;
    const unsigned short* s = packedW + (((size_t)ntile * KT + kt) * 32 + l) * 16;
    unsigned short*       d = smem    + ((size_t)tl * 32 + l) * 16;
    *(u32x4*)(d)     = *(const u32x4*)(s);
    *(u32x4*)(d + 8) = *(const u32x4*)(s + 8);
  }
  __syncthreads();

  // biases are time-invariant: hoist
  float bias[4];
#pragma unroll
  for (int g = 0; g < 4; ++g) bias[g] = bih[g * 512 + col] + bhh[g * 512 + col];

  // per-lane LDS base; fragment (g,kt) at + (g*KT + kt)*512 elements (const offsets)
  const unsigned short* sbase = smem + (size_t)lane * 16;

  v8f cst = (v8f){0.f,0.f,0.f,0.f,0.f,0.f,0.f,0.f};

  unsigned short* hb0 = hbuf;
  unsigned short* hb1 = hbuf + 64 * 512;

  for (int t = 0; t < T; ++t) {
    const unsigned short* hcur = (t & 1) ? hb1 : hb0;
    unsigned short*       hnxt = (t & 1) ? hb0 : hb1;

    // A bases (already include per-lane +half*8 elements)
    const unsigned short* ax = xin + ((size_t)(t * 64 + arow)) * IN + half * 8;
    const unsigned short* ah = hcur + (size_t)arow * 512 + half * 8;
    auto aptr = [&](int kt) -> const unsigned short* {
      return (kt < KIN) ? (ax + kt * 32) : (ah + (kt - KIN) * 32);
    };

    v8f acc[4];
#pragma unroll
    for (int g = 0; g < 4; ++g) acc[g] = (v8f){0.f,0.f,0.f,0.f,0.f,0.f,0.f,0.f};

    // A: depth-2 global prefetch (3-slot rotation avoids WAR on in-use slot)
    v16h abuf[3];
    abuf[0] = ldfragA(aptr(0));
    abuf[1] = ldfragA(aptr(1));
    // B: depth-1 double buffer from LDS
    v16h bbuf[2][4];
#pragma unroll
    for (int g = 0; g < 4; ++g) bbuf[0][g] = ldfragB_lds(sbase + (size_t)g * KT * 512);

#pragma unroll
    for (int kt = 0; kt < KT; ++kt) {
      const int cur = kt & 1, nxt = cur ^ 1;
      if (kt + 2 < KT) abuf[(kt + 2) % 3] = ldfragA(aptr(kt + 2));
      if (kt + 1 < KT) {
#pragma unroll
        for (int g = 0; g < 4; ++g)
          bbuf[nxt][g] = ldfragB_lds(sbase + ((size_t)g * KT + (kt + 1)) * 512);
      }
#pragma unroll
      for (int g = 0; g < 4; ++g)
        acc[g] = __builtin_amdgcn_wmma_f32_16x16x32_f16(
            false, abuf[kt % 3], false, bbuf[cur][g], (short)0, acc[g], false, false);
    }

    // cell update in fragment layout (no cross-lane traffic)
#pragma unroll
    for (int r = 0; r < 8; ++r) {
      float iv = sigmoidf(acc[0][r] + bias[0]);
      float fv = sigmoidf(acc[1][r] + bias[1]);
      float gv = tanhf   (acc[2][r] + bias[2]);
      float ov = sigmoidf(acc[3][r] + bias[3]);
      float cv = fv * cst[r] + iv * gv;
      cst[r] = cv;
      float hv = ov * tanhf(cv);
      int drow = m * 16 + r + 8 * half;           // D-layout row
      unsigned short h16 = f2h(hv);
      hnxt[(size_t)drow * 512 + col] = h16;
      hseq[((size_t)t * 64 + drow) * 512 + col] = h16;
      if (t == T - 1) hT[(size_t)drow * 512 + col] = hv;
    }

    // device-wide barrier across the 32 resident WGs (double buffer -> 1 sync/step)
    __syncthreads();
    if (tid == 0) {
      __threadfence();
      atomicAdd(cnt, 1u);
      unsigned target = gridDim.x * (unsigned)(t + 1);
      while (atomicAdd(cnt, 0u) < target) { __builtin_amdgcn_s_sleep(1); }
    }
    __syncthreads();
  }
}

// ---------- tiny FC: out[b][n] = hT[b] . W_fc[n] + b_fc[n] ----------
__global__ void fc_kernel(const float* __restrict__ hT, const float* __restrict__ Wfc,
                          const float* __restrict__ bfc, float* __restrict__ out) {
  int tid = blockIdx.x * blockDim.x + threadIdx.x;
  if (tid >= 64 * 5) return;
  int b = tid / 5, n = tid % 5;
  float s = bfc[n];
  for (int k = 0; k < 512; ++k) s += hT[(size_t)b * 512 + k] * Wfc[(size_t)n * 512 + k];
  out[(size_t)b * 5 + n] = s;
}

extern "C" void kernel_launch(void* const* d_in, const int* in_sizes, int n_in,
                              void* d_out, int out_size, void* d_ws, size_t ws_size,
                              hipStream_t stream) {
  const float* x    = (const float*)d_in[0];  // [64,64,1000]
  const float* Wih0 = (const float*)d_in[1];  // [2048,64]
  const float* WihR = (const float*)d_in[2];  // [2,2048,512]
  const float* Whh  = (const float*)d_in[3];  // [3,2048,512]
  const float* bih  = (const float*)d_in[4];  // [3,2048]
  const float* bhh  = (const float*)d_in[5];  // [3,2048]
  const float* Wfc  = (const float*)d_in[6];  // [5,512]
  const float* bfc  = (const float*)d_in[7];  // [5]
  float* out = (float*)d_out;                 // [64,5]

  char* ws = (char*)d_ws;
  size_t o = 0;
  auto alloc = [&](size_t bytes) -> char* {
    char* p = ws + o;
    o = (o + bytes + 255) & ~(size_t)255;
    return p;
  };
  unsigned*       cnt  = (unsigned*)      alloc(4);
  unsigned short* hbuf = (unsigned short*)alloc((size_t)2 * 64 * 512 * 2);
  float*          hT   = (float*)         alloc((size_t)64 * 512 * 4);
  unsigned short* xin0 = (unsigned short*)alloc((size_t)64000 * 64 * 2);
  unsigned short* seqA = (unsigned short*)alloc((size_t)64000 * 512 * 2);
  unsigned short* seqB = (unsigned short*)alloc((size_t)64000 * 512 * 2);
  unsigned short* pk0  = (unsigned short*)alloc((size_t)128 * 18 * 1024);
  unsigned short* pk1  = (unsigned short*)alloc((size_t)128 * 32 * 1024);
  unsigned short* pk2  = (unsigned short*)alloc((size_t)128 * 32 * 1024);

  // one-time (per call) weight packing + input conversion
  pack_weights<<<(128 * 18 * 32 + 255) / 256, 256, 0, stream>>>(Wih0, Whh, 64, 18, pk0);
  pack_weights<<<(128 * 32 * 32 + 255) / 256, 256, 0, stream>>>(
      WihR, Whh + (size_t)2048 * 512, 512, 32, pk1);
  pack_weights<<<(128 * 32 * 32 + 255) / 256, 256, 0, stream>>>(
      WihR + (size_t)2048 * 512, Whh + (size_t)2 * 2048 * 512, 512, 32, pk2);
  x_to_f16<<<(64 * 64 * 1000 + 255) / 256, 256, 0, stream>>>(x, xin0);

  // layer 0: K = 64 + 512 (KIN = 2 tiles)
  scan_init<<<128, 256, 0, stream>>>((unsigned*)hbuf, cnt);
  lstm_scan<2><<<32, 128, 0, stream>>>(xin0, pk0, bih + 0, bhh + 0,
                                       seqA, hbuf, hT, cnt, 1000);
  // layer 1: K = 512 + 512 (KIN = 16 tiles)
  scan_init<<<128, 256, 0, stream>>>((unsigned*)hbuf, cnt);
  lstm_scan<16><<<32, 128, 0, stream>>>(seqA, pk1, bih + 2048, bhh + 2048,
                                        seqB, hbuf, hT, cnt, 1000);
  // layer 2
  scan_init<<<128, 256, 0, stream>>>((unsigned*)hbuf, cnt);
  lstm_scan<16><<<32, 128, 0, stream>>>(seqB, pk2, bih + 4096, bhh + 4096,
                                        seqA, hbuf, hT, cnt, 1000);

  fc_kernel<<<2, 160, 0, stream>>>(hT, Wfc, bfc, out);
}
